// unit_gcn_38560216383770
// MI455X (gfx1250) — compile-verified, matrix operand order checked by reading, not run on hardware
//
#include <hip/hip_runtime.h>

typedef _Float16 f16;
typedef __attribute__((ext_vector_type(8)))  _Float16 v8h;
typedef __attribute__((ext_vector_type(16))) _Float16 v16h;
typedef __attribute__((ext_vector_type(8)))  float    v8f;

// ---- problem constants (from reference) ----
#define N_BATCH 64
#define C_IN    64
#define C_OUT   64
#define K_SUB   3
#define O_DIM   192           // C_OUT * K_SUB
#define T_DIM   256
#define V_DIM   25
#define T_TILE  4
#define S_DIM   100           // T_TILE * V_DIM
#define S_PAD   112           // 7 tiles of 16
#define N_STRIDE 409600       // C_IN*T_DIM*V_DIM
#define C_STRIDE 6400         // T_DIM*V_DIM

// ---- LDS layout (bytes) ----
#define WROW 72               // padded row length (f16) for W and Xs
#define YROW 40               // padded v-length (f16) for Ys
#define AROW 104              // padded K-length (f16) for stacked A
#define WS_OFF    0                        // f16 [192][72]  = 27648 B
#define XS_OFF    27648                    // f16 [112][72]  = 16128 B
#define AS_OFF    43776                    // f16 [32][104]  =  6656 B
#define YS_OFF    50432                    // f16 [4][3][64][40] = 61440 B
#define BIAS_OFF  111872                   // f32 [192] = 768 B
#define SCALE_OFF 112640                   // f32 [64]
#define SHIFT_OFF 112896                   // f32 [64]
#define SMEM_TOTAL 113152

__device__ __forceinline__ v16h load_frag(const f16* p) {
  // two 16B chunks: K = {base..base+7} and {base+16..base+23} per ISA layout
  v8h lo = *(const v8h*)(p);
  v8h hi = *(const v8h*)(p + 16);
  return __builtin_shufflevector(lo, hi, 0,1,2,3,4,5,6,7,8,9,10,11,12,13,14,15);
}

__global__ __launch_bounds__(256) void unit_gcn_fused_kernel(
    const float* __restrict__ xg, const float* __restrict__ Ag,
    const float* __restrict__ wg, const float* __restrict__ bg,
    const float* __restrict__ gamma, const float* __restrict__ beta,
    const float* __restrict__ rmean, const float* __restrict__ rvar,
    float* __restrict__ out) {
  extern __shared__ char smem[];
  f16*  ws    = (f16*)(smem + WS_OFF);
  f16*  xs    = (f16*)(smem + XS_OFF);
  f16*  as_   = (f16*)(smem + AS_OFF);
  f16*  ys    = (f16*)(smem + YS_OFF);
  float* biasF  = (float*)(smem + BIAS_OFF);
  float* scaleF = (float*)(smem + SCALE_OFF);
  float* shiftF = (float*)(smem + SHIFT_OFF);

  const int tid  = threadIdx.x;
  const int wave = tid >> 5;
  const int lane = tid & 31;
  const int n    = blockIdx.y;
  const int t0   = blockIdx.x * T_TILE;

  // ---------------- Phase 0a: zero padded LDS regions (Xs, As, Ys are contiguous)
  {
    unsigned* z = (unsigned*)(smem + XS_OFF);
    const int ndw = (YS_OFF + 61440 - XS_OFF) >> 2;   // 21056 dwords
    for (int i = tid; i < ndw; i += 256) z[i] = 0u;
  }
  __syncthreads();

  // ---------------- Phase 0b: stage operands
  // conv_w (192x64 fp32, row-major) -> ws[o][c] f16, padded rows
  for (int i = tid; i < O_DIM * C_IN; i += 256) {
    int o = i >> 6, c = i & 63;
    ws[o * WROW + c] = (f16)wg[i];
  }
  // x slab -> xs[s][c] f16 (transpose: global is c-major, LDS is s-major)
  {
    const long xbase = (long)n * N_STRIDE + (long)t0 * V_DIM;
    for (int i = tid; i < C_IN * S_DIM; i += 256) {
      int c = i / S_DIM, s = i - c * S_DIM;
      xs[s * WROW + c] = (f16)xg[xbase + (long)c * C_STRIDE + s];
    }
  }
  // A (3x25x25) -> as_[w][k*32+v] f16 (stacked, K padded to 96 of 104)
  for (int i = tid; i < K_SUB * V_DIM * V_DIM; i += 256) {
    int kk = i / 625, r = i - kk * 625;
    int v = r / V_DIM, w = r - v * V_DIM;
    as_[w * AROW + kk * 32 + v] = (f16)Ag[i];
  }
  if (tid < O_DIM) biasF[tid] = bg[tid];
  if (tid < C_OUT) {
    float sc = gamma[tid] * rsqrtf(rvar[tid] + 1e-5f);
    scaleF[tid] = sc;
    shiftF[tid] = beta[tid] - rmean[tid] * sc;
  }
  __syncthreads();

  const int lrow = lane & 15;
  const int half = (lane >> 4) << 3;          // 0 for lanes 0-15, 8 for 16-31
  const int moff = (lane >> 4) << 3;          // C/D: M = r (+8 for upper lanes)

  // ---------------- Phase 1: Y[192 x 112] = W x X  (K=64, 2 WMMA steps)
  for (int q = wave; q < 12 * 7; q += 8) {
    const int mt = q / 7, nt = q - mt * 7;
    const int m0 = mt * 16, s0 = nt * 16;

    v8f acc;
    const int mbase = m0 + moff;
    #pragma unroll
    for (int r = 0; r < 8; ++r) acc[r] = biasF[mbase + r];   // fold conv bias into C

    const f16* wrow = ws + (m0 + lrow) * WROW + half;
    const f16* xrow = xs + (s0 + lrow) * WROW + half;

    v16h a0 = load_frag(wrow);
    v16h b0 = load_frag(xrow);
    acc = __builtin_amdgcn_wmma_f32_16x16x32_f16(false, a0, false, b0, (short)0, acc, false, false);
    v16h a1 = load_frag(wrow + 32);
    v16h b1 = load_frag(xrow + 32);
    acc = __builtin_amdgcn_wmma_f32_16x16x32_f16(false, a1, false, b1, (short)0, acc, false, false);

    const int s = s0 + lrow;
    if (s < S_DIM) {
      const int ts = s / V_DIM, v = s - ts * V_DIM;
      #pragma unroll
      for (int r = 0; r < 8; ++r) {
        int o = mbase + r;
        int kk = o >> 6, c = o & 63;
        ys[((ts * K_SUB + kk) * C_OUT + c) * YROW + v] = (f16)acc[r];
      }
    }
  }
  __syncthreads();

  // ---------------- Phase 2: Z[64 x 32] = Y[c,(k,v)] x Astk[(k,v),w]  (K=96, 3 steps)
  for (int q = wave; q < 4 * 4 * 2; q += 8) {
    const int t  = q >> 3;
    const int mt = (q >> 1) & 3;
    const int nt = q & 1;
    const int m0 = mt * 16, w0 = nt * 16;

    v8f acc = (v8f)(0.0f);
    const f16* arow = as_ + (w0 + lrow) * AROW + half;
    #pragma unroll
    for (int kk = 0; kk < K_SUB; ++kk) {
      v16h a = load_frag(ys + ((t * K_SUB + kk) * C_OUT + m0 + lrow) * YROW + half);
      v16h b = load_frag(arow + kk * 32);
      acc = __builtin_amdgcn_wmma_f32_16x16x32_f16(false, a, false, b, (short)0, acc, false, false);
    }

    const int w = w0 + lrow;
    if (w < V_DIM) {
      const int cbase = m0 + moff;
      const int tg = t0 + t;
      const long base = (long)n * N_STRIDE + (long)tg * V_DIM + w;
      #pragma unroll
      for (int r = 0; r < 8; ++r) {
        int c = cbase + r;
        long idx = base + (long)c * C_STRIDE;
        float z = acc[r] * scaleF[c] + shiftF[c] + xg[idx];   // BN + residual
        out[idx] = z > 0.0f ? z : 0.0f;                       // ReLU
      }
    }
  }
}

extern "C" void kernel_launch(void* const* d_in, const int* in_sizes, int n_in,
                              void* d_out, int out_size, void* d_ws, size_t ws_size,
                              hipStream_t stream) {
  const float* x      = (const float*)d_in[0];
  const float* A      = (const float*)d_in[1];
  const float* conv_w = (const float*)d_in[2];
  const float* conv_b = (const float*)d_in[3];
  const float* gamma  = (const float*)d_in[4];
  const float* beta   = (const float*)d_in[5];
  const float* rmean  = (const float*)d_in[6];
  const float* rvar   = (const float*)d_in[7];
  float* out = (float*)d_out;

  dim3 grid(T_DIM / T_TILE, N_BATCH);   // 64 x 64 blocks
  unit_gcn_fused_kernel<<<grid, 256, SMEM_TOTAL, stream>>>(
      x, A, conv_w, conv_b, gamma, beta, rmean, rvar, out);
}
